// nade_44959717655093
// MI455X (gfx1250) — compile-verified, matrix operand order checked by reading, not run on hardware
//
#include <hip/hip_runtime.h>
#include <hip/hip_bf16.h>

// NADE forward, fused. B=64, N=4096, N_H=128.
// Phase 1: chunk-carry GEMM via v_wmma_f32_16x16x32_bf16 (f32 accumulate),
//          operands fetched as b128 vector loads per the ISA lane layout.
// Phase 2: fused scan + sigmoid + W2 reduction + output sigmoid (dominant).

#define NN   4096
#define BB   64
#define NH   128
#define CSZ  64          // chunk size along N
#define NC   (NN / CSZ)  // 64 chunks

typedef __attribute__((ext_vector_type(16))) __bf16 v16bf;
typedef __attribute__((ext_vector_type(8)))  float  v8f;

__device__ __forceinline__ __bf16 f2bf(float f) {
  unsigned u = __builtin_bit_cast(unsigned, f);
  u = (u + 0x7FFFu + ((u >> 16) & 1u)) >> 16;   // round-to-nearest-even
  unsigned short s = (unsigned short)u;
  return __builtin_bit_cast(__bf16, s);
}

__device__ __forceinline__ float fast_sigmoid(float t) {
  // sigmoid(t) = 1 / (1 + 2^(-t*log2(e)))  -> v_exp_f32 + v_rcp_f32
  float e = __builtin_amdgcn_exp2f(t * -1.44269504088896341f);
  return __builtin_amdgcn_rcpf(1.0f + e);
}

// Pack one lane's 16 bf16 operand elements from two contiguous float4 pairs.
// ISA 16-bit A/B layout: lane (half=lane>>4) holds K = half*8+{0..7} in the
// low 8 elements and K = 16+half*8+{0..7} in the high 8 elements.
__device__ __forceinline__ v16bf pack16(const float* __restrict__ row, int k0) {
  const float4 p0 = *(const float4*)(row + k0);
  const float4 p1 = *(const float4*)(row + k0 + 4);
  const float4 p2 = *(const float4*)(row + k0 + 16);
  const float4 p3 = *(const float4*)(row + k0 + 20);
  v16bf v;
  v[0]  = f2bf(p0.x); v[1]  = f2bf(p0.y); v[2]  = f2bf(p0.z); v[3]  = f2bf(p0.w);
  v[4]  = f2bf(p1.x); v[5]  = f2bf(p1.y); v[6]  = f2bf(p1.z); v[7]  = f2bf(p1.w);
  v[8]  = f2bf(p2.x); v[9]  = f2bf(p2.y); v[10] = f2bf(p2.z); v[11] = f2bf(p2.w);
  v[12] = f2bf(p3.x); v[13] = f2bf(p3.y); v[14] = f2bf(p3.z); v[15] = f2bf(p3.w);
  return v;
}

// ---------------------------------------------------------------------------
// Phase 1: P[c][b][i] = sum_{j in chunk c} x[b,j] * W1[i,j]
// One wave per 16x16 output tile. grid = (NC, 4 m-tiles, 8 n-tiles).
// ---------------------------------------------------------------------------
__global__ __launch_bounds__(32) void nade_phase1_wmma(
    const float* __restrict__ x, const float* __restrict__ W1,
    float* __restrict__ P)
{
  const int c    = blockIdx.x;   // chunk
  const int tm   = blockIdx.y;   // b-tile 0..3
  const int tn   = blockIdx.z;   // i-tile 0..7
  const int lane = threadIdx.x;  // 0..31 (wave32)
  const int half = lane >> 4;    // 0: K {0-7,16-23}, 1: K {8-15,24-31}
  const int mn   = lane & 15;    // M index (A) / N index (B)

  const float* __restrict__ xrow = x  + (tm * 16 + mn) * NN;  // A: M across lanes
  const float* __restrict__ wrow = W1 + (tn * 16 + mn) * NN;  // B: N across lanes

  v8f acc = {};
#pragma unroll
  for (int kk = 0; kk < 2; ++kk) {  // 64-wide chunk = 2 x K=32
    const int k0 = c * CSZ + kk * 32 + half * 8;
    const v16bf a  = pack16(xrow, k0);
    const v16bf bm = pack16(wrow, k0);
    acc = __builtin_amdgcn_wmma_f32_16x16x32_bf16(
        /*neg_a=*/false, a, /*neg_b=*/false, bm,
        /*c_mod=*/(short)0, acc, /*reuse_a=*/false, /*reuse_b=*/false);
  }

  // D layout: lane column N = mn (+tn*16); VGPR r -> row M = r + 8*half
#pragma unroll
  for (int r = 0; r < 8; ++r) {
    const int brow = tm * 16 + r + 8 * half;
    P[(c * BB + brow) * NH + tn * 16 + mn] = acc[r];
  }
}

// ---------------------------------------------------------------------------
// Exclusive scan over chunks (in place): P[c][b][i] <- sum_{c'<c} P[c'][b][i]
// One thread per (b,i) line; reads/writes coalesced at each chunk step.
// ---------------------------------------------------------------------------
__global__ __launch_bounds__(128) void nade_scan(float* __restrict__ P)
{
  const int t = blockIdx.x * 128 + threadIdx.x;   // 0..8191
  const int b = t >> 7;
  const int i = t & (NH - 1);
  float acc = 0.0f;
#pragma unroll 4
  for (int c = 0; c < NC; ++c) {
    const int idx = (c * BB + b) * NH + i;
    const float v = P[idx];
    P[idx] = acc;
    acc += v;
  }
}

// ---------------------------------------------------------------------------
// Phase 2 (dominant): one block = (b, chunk c), 128 threads = i dimension.
// Thread i: exclusive in-chunk scan seeded by carry, sigmoid, stage
// W2[i,j]*sig into stride-129 LDS (bank-conflict free), then reduce over i.
// ---------------------------------------------------------------------------
__global__ __launch_bounds__(128) void nade_phase2(
    const float* __restrict__ x,  const float* __restrict__ W1,
    const float* __restrict__ b1, const float* __restrict__ W2,
    const float* __restrict__ b2, const float* __restrict__ S,
    float* __restrict__ out)
{
  const int c   = blockIdx.x;
  const int b   = blockIdx.y;
  const int tid = threadIdx.x;        // == i
  const int j0  = c * CSZ;

  __shared__ __align__(16) float xs[CSZ];
  __shared__ float contrib[CSZ * (NH + 1)];   // [jj][i], stride 129
  __shared__ float psum[128];

  if (tid < CSZ) xs[tid] = x[b * NN + j0 + tid];
  __syncthreads();

  const int   i   = tid;
  float       acc = S[(c * BB + b) * NH + i];   // carry-in: sum_{j < j0}
  const float bi  = b1[i];
  const float4* __restrict__ w1q = (const float4*)(W1 + i * NN + j0);
  const float4* __restrict__ w2q = (const float4*)(W2 + i * NN + j0);
  const float4* __restrict__ xq  = (const float4*)xs;

  __builtin_prefetch(w1q, 0, 3);
  __builtin_prefetch(w2q, 0, 3);

#pragma unroll
  for (int q = 0; q < CSZ / 4; ++q) {
    const float4 w1v = w1q[q];
    const float4 w2v = w2q[q];
    const float4 xv  = xq[q];
    float* cp = contrib + (q * 4) * (NH + 1) + i;
    // exclusive: sigmoid uses acc BEFORE adding the current column's product
    float sg;
    sg = fast_sigmoid(acc + bi); cp[0 * (NH + 1)] = w2v.x * sg; acc = fmaf(w1v.x, xv.x, acc);
    sg = fast_sigmoid(acc + bi); cp[1 * (NH + 1)] = w2v.y * sg; acc = fmaf(w1v.y, xv.y, acc);
    sg = fast_sigmoid(acc + bi); cp[2 * (NH + 1)] = w2v.z * sg; acc = fmaf(w1v.z, xv.z, acc);
    sg = fast_sigmoid(acc + bi); cp[3 * (NH + 1)] = w2v.w * sg; acc = fmaf(w1v.w, xv.w, acc);
  }
  __syncthreads();

  // Reduce over i: 128 threads -> 64 columns x 2 half-sums each.
  const int jj = tid & 63;
  const int hh = tid >> 6;
  const float* rp = contrib + jj * (NH + 1) + hh * 64;
  float s = 0.0f;
#pragma unroll 8
  for (int k = 0; k < 64; ++k) s += rp[k];
  psum[tid] = s;
  __syncthreads();

  if (tid < CSZ) {
    const float yv = psum[tid] + psum[tid + 64] + b2[j0 + tid];
    out[b * NN + j0 + tid] = fast_sigmoid(yv);
  }
}

// ---------------------------------------------------------------------------
extern "C" void kernel_launch(void* const* d_in, const int* in_sizes, int n_in,
                              void* d_out, int out_size, void* d_ws, size_t ws_size,
                              hipStream_t stream) {
  const float* x  = (const float*)d_in[0];
  const float* W1 = (const float*)d_in[1];
  const float* b1 = (const float*)d_in[2];
  const float* W2 = (const float*)d_in[3];
  const float* b2 = (const float*)d_in[4];
  float* out = (float*)d_out;
  float* P   = (float*)d_ws;   // NC*BB*NH floats = 2 MB of scratch

  dim3 g1(NC, BB / 16, NH / 16);          // 64 x 4 x 8 waves (one 16x16 tile each)
  nade_phase1_wmma<<<g1, 32, 0, stream>>>(x, W1, P);

  nade_scan<<<(BB * NH) / 128, 128, 0, stream>>>(P);

  dim3 g2(NC, BB);                        // 4096 blocks x 4 waves
  nade_phase2<<<g2, 128, 0, stream>>>(x, W1, b1, W2, b2, P, out);
}
